// ConditionModulationBlock_55508157333802
// MI455X (gfx1250) — matrix operand order repaired and checked by reading, not compiled, not run on hardware
//
#include <hip/hip_runtime.h>
#include <hip/hip_bf16.h>

typedef __attribute__((ext_vector_type(16))) _Float16 v16h;
typedef __attribute__((ext_vector_type(8)))  _Float16 v8h;
typedef __attribute__((ext_vector_type(8)))  float    v8f;

#define D_MODEL 1024
#define D_INNER 2048
#define D_STATE 16
#define DT_RANK 64
#define XPO     96          // DT_RANK + 2*D_STATE
#define BATCH   2
#define SEQ     1024
#define BL      (BATCH*SEQ) // 2048

// ---------------------------------------------------------------------------
// f32 -> f16 conversion (weights)
// ---------------------------------------------------------------------------
__global__ void k_cvt(const float* __restrict__ s, _Float16* __restrict__ d, int n) {
  int i = blockIdx.x * blockDim.x + threadIdx.x;
  if (i < n) d[i] = (_Float16)s[i];
}

// ---------------------------------------------------------------------------
// fused residual add + LayerNorm; writes residual (f32, output #2) and
// normed activations in f16 (GEMM A operand)
// ---------------------------------------------------------------------------
__global__ __launch_bounds__(256) void k_add_ln(
    const float* __restrict__ hs, const float* __restrict__ res_in,
    const float* __restrict__ w, const float* __restrict__ bvec,
    float* __restrict__ res_out, _Float16* __restrict__ normed) {
  int row = blockIdx.x;                 // BL rows of D_MODEL
  int tid = threadIdx.x;
  __shared__ float ssum[256];
  __shared__ float ssq[256];
  const float* h = hs     + (size_t)row * D_MODEL;
  const float* r = res_in + (size_t)row * D_MODEL;
  float v[4], s = 0.f, q = 0.f;
#pragma unroll
  for (int i = 0; i < 4; i++) {
    int d = tid * 4 + i;
    v[i] = h[d] + r[d];
    s += v[i]; q += v[i] * v[i];
  }
  ssum[tid] = s; ssq[tid] = q;
  __syncthreads();
  for (int off = 128; off > 0; off >>= 1) {
    if (tid < off) { ssum[tid] += ssum[tid + off]; ssq[tid] += ssq[tid + off]; }
    __syncthreads();
  }
  float mean = ssum[0] * (1.f / D_MODEL);
  float var  = ssq[0]  * (1.f / D_MODEL) - mean * mean;
  float rstd = rsqrtf(var + 1e-5f);
  float* ro = res_out + (size_t)row * D_MODEL;
  _Float16* no = normed + (size_t)row * D_MODEL;
#pragma unroll
  for (int i = 0; i < 4; i++) {
    int d = tid * 4 + i;
    ro[d] = v[i];
    no[d] = (_Float16)((v[i] - mean) * rstd * w[d] + bvec[d]);
  }
}

// ---------------------------------------------------------------------------
// WMMA GEMM: C[M,N] (f32) = A[M,K] (f16, row-major) x W[N,K]^T (f16, row-major)
// wave32: each wave owns a 32(M) x 64(N) strip -> 8 f32 accumulators; the B
// tile is reused across both A row-blocks (2x the MACs per byte vs 16x64).
// 8 waves / block -> 256x64 of C per block. K must be a multiple of 32.
// Per CDNA5 ISA layouts:
//   A lane(j,h): row m0+j, halves [k+8h .. +7] and [k+16+8h .. +7]
//   B lane(j,h): W row n0+j (== column of B), halves [k+8h .. +7] and [k+16+8h .. +7]
//   C lane(j,h), vgpr r: C[m0 + r + 8h][n0 + j]
// Epilogues: 0 = plain f32 store
//            1 = +bias[col], softplus (dt_proj)
//            2 = f32 store + f16 aux store for cols < DT_RANK (x_proj)
// ---------------------------------------------------------------------------
template <int EPI>
__global__ __launch_bounds__(256) void k_gemm_f16(
    const _Float16* __restrict__ A, const _Float16* __restrict__ W,
    float* __restrict__ C, int M, int N, int K,
    const float* __restrict__ bias, _Float16* __restrict__ aux) {
  int tid  = threadIdx.x;
  int wv   = tid >> 5;
  int lane = tid & 31;
  int j  = lane & 15;
  int hh = lane >> 4;
  int mt = blockIdx.y * 256 + wv * 32;
  int nt = blockIdx.x * 64;

  int mrow0 = mt + j;      if (mrow0 >= M) mrow0 = M - 1;  // clamp: EXEC all-1s
  int mrow1 = mt + 16 + j; if (mrow1 >= M) mrow1 = M - 1;
  const _Float16* arow0 = A + (size_t)mrow0 * K;
  const _Float16* arow1 = A + (size_t)mrow1 * K;

  v8f acc[2][4];
#pragma unroll
  for (int u = 0; u < 2; u++)
#pragma unroll
    for (int t = 0; t < 4; t++)
      acc[u][t] = (v8f){0.f,0.f,0.f,0.f,0.f,0.f,0.f,0.f};

  for (int kt = 0; kt < K; kt += 32) {
    __builtin_prefetch(arow0 + kt + 512, 0, 0);
    __builtin_prefetch(arow1 + kt + 512, 0, 0);
    v8h a0lo = *(const v8h*)(arow0 + kt + 8 * hh);
    v8h a0hi = *(const v8h*)(arow0 + kt + 16 + 8 * hh);
    v8h a1lo = *(const v8h*)(arow1 + kt + 8 * hh);
    v8h a1hi = *(const v8h*)(arow1 + kt + 16 + 8 * hh);
    v16h a0, a1;
#pragma unroll
    for (int i = 0; i < 8; i++) {
      a0[i] = a0lo[i]; a0[i + 8] = a0hi[i];
      a1[i] = a1lo[i]; a1[i + 8] = a1hi[i];
    }
#pragma unroll
    for (int t = 0; t < 4; t++) {
      int nn = nt + t * 16 + j; if (nn >= N) nn = N - 1;
      v8h blo = *(const v8h*)(W + (size_t)nn * K + kt + 8 * hh);
      v8h bhi = *(const v8h*)(W + (size_t)nn * K + kt + 16 + 8 * hh);
      v16h b;
#pragma unroll
      for (int i = 0; i < 8; i++) { b[i] = blo[i]; b[i + 8] = bhi[i]; }
      acc[0][t] = __builtin_amdgcn_wmma_f32_16x16x32_f16(
          false, a0, false, b, (short)0, acc[0][t], false, false);
      acc[1][t] = __builtin_amdgcn_wmma_f32_16x16x32_f16(
          false, a1, false, b, (short)0, acc[1][t], false, false);
    }
  }
#pragma unroll
  for (int u = 0; u < 2; u++) {
#pragma unroll
    for (int t = 0; t < 4; t++) {
      int nn = nt + t * 16 + j;
      if (nn < N) {
#pragma unroll
        for (int r = 0; r < 8; r++) {
          int mr = mt + u * 16 + r + 8 * hh;
          if (mr < M) {
            float x = acc[u][t][r];
            if (EPI == 1) {                       // dt_proj: bias + softplus
              x += bias[nn];
              x = (x > 20.f) ? x : log1pf(__expf(x));
            }
            C[(size_t)mr * N + nn] = x;
            if (EPI == 2 && nn < DT_RANK)         // x_proj: dt slice -> f16
              aux[(size_t)mr * DT_RANK + nn] = (_Float16)x;
          }
        }
      }
    }
  }
}

// ---------------------------------------------------------------------------
// depthwise causal conv (k=4) over L + bias + SiLU; emits xc in f32 and f16
// ---------------------------------------------------------------------------
__global__ __launch_bounds__(256) void k_conv_silu(
    const float* __restrict__ xz, const float* __restrict__ cw,
    const float* __restrict__ cb, float* __restrict__ xcf,
    _Float16* __restrict__ xch) {
  int idx = blockIdx.x * blockDim.x + threadIdx.x;  // B*L*ED
  int e = idx & (D_INNER - 1);
  int l = (idx >> 11) & (SEQ - 1);
  int b = idx >> 21;
  float acc = cb[e];
#pragma unroll
  for (int jj = 0; jj < 4; jj++) {
    int ls = l + jj - 3;
    if (ls >= 0)
      acc += xz[((size_t)(b * SEQ + ls)) * (2 * D_INNER) + e] * cw[e * 4 + jj];
  }
  float s = acc / (1.f + __expf(-acc));
  xcf[idx] = s;
  xch[idx] = (_Float16)s;
}

// ---------------------------------------------------------------------------
// selective scan: one thread per (b, channel). h[16] held in registers across
// the whole L=1024 recurrence. B/C (shared across channels) staged in LDS in
// chunks of 32 timesteps. Fused: +xc*D, *silu(z), f16 emit for out_proj.
// ---------------------------------------------------------------------------
__global__ __launch_bounds__(256) void k_scan(
    const float* __restrict__ delta, const float* __restrict__ xc,
    const float* __restrict__ xz,    const float* __restrict__ xdbl,
    const float* __restrict__ A_log, const float* __restrict__ Dp,
    _Float16* __restrict__ yf16) {
  int tid = threadIdx.x;
  int b = blockIdx.x >> 3;                       // 8 blocks of 256 ch per batch
  int e = ((blockIdx.x & 7) << 8) + tid;

  float Aneg[D_STATE], hst[D_STATE];
#pragma unroll
  for (int n = 0; n < D_STATE; n++) {
    Aneg[n] = -__expf(A_log[e * D_STATE + n]);
    hst[n] = 0.f;
  }
  float Dv = Dp[e];

  __shared__ float sBC[32][32];                  // [timestep][Bm0..15|Cm0..15]
  for (int l0 = 0; l0 < SEQ; l0 += 32) {
    __syncthreads();
#pragma unroll
    for (int i = 0; i < 4; i++) {
      int idx = tid * 4 + i;                     // 1024 floats / chunk
      int cl = idx >> 5, c = idx & 31;
      sBC[cl][c] = xdbl[((size_t)(b * SEQ + l0 + cl)) * XPO + DT_RANK + c];
    }
    __syncthreads();
    for (int cl = 0; cl < 32; cl++) {
      size_t row = (size_t)(b * SEQ + l0 + cl);
      float dv = delta[row * D_INNER + e];
      float xv = xc[row * D_INNER + e];
      float zv = xz[row * (2 * D_INNER) + D_INNER + e];
      float y = 0.f;
#pragma unroll
      for (int n = 0; n < D_STATE; n++) {
        float dA = __expf(dv * Aneg[n]);
        hst[n] = dA * hst[n] + (dv * sBC[cl][n]) * xv;
        y += hst[n] * sBC[cl][16 + n];
      }
      y += xv * Dv;
      float sig = 1.f / (1.f + __expf(-zv));
      y *= zv * sig;
      yf16[row * D_INNER + e] = (_Float16)y;
    }
  }
}

// ---------------------------------------------------------------------------
extern "C" void kernel_launch(void* const* d_in, const int* in_sizes, int n_in,
                              void* d_out, int out_size, void* d_ws, size_t ws_size,
                              hipStream_t stream) {
  const float* hs     = (const float*)d_in[0];
  const float* res    = (const float*)d_in[1];
  const float* norm_w = (const float*)d_in[2];
  const float* norm_b = (const float*)d_in[3];
  const float* w_in   = (const float*)d_in[4];   // [4096,1024]
  const float* conv_w = (const float*)d_in[5];   // [2048,1,4]
  const float* conv_b = (const float*)d_in[6];
  const float* w_xp   = (const float*)d_in[7];   // [96,2048]
  const float* w_dt   = (const float*)d_in[8];   // [2048,64]
  const float* dt_b   = (const float*)d_in[9];
  const float* A_log  = (const float*)d_in[10];  // [2048,16]
  const float* Dp     = (const float*)d_in[11];
  const float* w_out  = (const float*)d_in[12];  // [1024,2048]

  float* out_y   = (float*)d_out;                          // [BL,1024]
  float* out_res = (float*)d_out + (size_t)BL * D_MODEL;   // [BL,1024]

  char* ws = (char*)d_ws;
  size_t o = 0;
  auto alloc = [&](size_t bytes) { char* p = ws + o; o += (bytes + 255) & ~(size_t)255; return p; };
  _Float16* normed_h = (_Float16*)alloc((size_t)BL * D_MODEL * 2);
  _Float16* w_in_h   = (_Float16*)alloc((size_t)(2*D_INNER) * D_MODEL * 2);
  _Float16* w_xp_h   = (_Float16*)alloc((size_t)XPO * D_INNER * 2);
  _Float16* w_dt_h   = (_Float16*)alloc((size_t)D_INNER * DT_RANK * 2);
  _Float16* w_out_h  = (_Float16*)alloc((size_t)D_MODEL * D_INNER * 2);
  float*    xzf      = (float*)   alloc((size_t)BL * 2 * D_INNER * 4);
  float*    xcf      = (float*)   alloc((size_t)BL * D_INNER * 4);
  _Float16* xch      = (_Float16*)alloc((size_t)BL * D_INNER * 2);
  float*    xdbl     = (float*)   alloc((size_t)BL * XPO * 4);
  _Float16* dth      = (_Float16*)alloc((size_t)BL * DT_RANK * 2);
  float*    deltaf   = (float*)   alloc((size_t)BL * D_INNER * 4);
  _Float16* yh       = (_Float16*)alloc((size_t)BL * D_INNER * 2);

  // 1) weights -> f16
  { int n = 2*D_INNER*D_MODEL; k_cvt<<<(n+255)/256,256,0,stream>>>(w_in,  w_in_h,  n); }
  { int n = XPO*D_INNER;       k_cvt<<<(n+255)/256,256,0,stream>>>(w_xp,  w_xp_h,  n); }
  { int n = D_INNER*DT_RANK;   k_cvt<<<(n+255)/256,256,0,stream>>>(w_dt,  w_dt_h,  n); }
  { int n = D_MODEL*D_INNER;   k_cvt<<<(n+255)/256,256,0,stream>>>(w_out, w_out_h, n); }

  // 2) add + LN (residual output + f16 activations)
  k_add_ln<<<BL, 256, 0, stream>>>(hs, res, norm_w, norm_b, out_res, normed_h);

  // 3) in_proj: [2048,1024] x [1024,4096] -> xz
  k_gemm_f16<0><<<dim3((2*D_INNER)/64, BL/256), 256, 0, stream>>>(
      normed_h, w_in_h, xzf, BL, 2*D_INNER, D_MODEL, nullptr, nullptr);

  // 4) causal depthwise conv + SiLU
  k_conv_silu<<<(BL*D_INNER)/256, 256, 0, stream>>>(xzf, conv_w, conv_b, xcf, xch);

  // 5) x_proj: [2048,2048] x [2048,96] -> x_dbl (+ dt slice f16 fused)
  k_gemm_f16<2><<<dim3((XPO+63)/64, BL/256), 256, 0, stream>>>(
      xch, w_xp_h, xdbl, BL, XPO, D_INNER, nullptr, dth);

  // 6) dt_proj: [2048,64] x [64,2048] -> delta (bias + softplus fused)
  k_gemm_f16<1><<<dim3(D_INNER/64, BL/256), 256, 0, stream>>>(
      dth, w_dt_h, deltaf, BL, D_INNER, DT_RANK, dt_b, nullptr);

  // 7) selective scan, fused skip + gate, f16 emit
  k_scan<<<(BATCH*D_INNER)/256, 256, 0, stream>>>(deltaf, xcf, xzf, xdbl, A_log, Dp, yh);

  // 8) out_proj: [2048,2048] x [2048,1024] -> out
  k_gemm_f16<0><<<dim3(D_MODEL/64, BL/256), 256, 0, stream>>>(
      yh, w_out_h, out_y, BL, D_MODEL, D_INNER, nullptr, nullptr);
}
